// ASAPBlock_56341380989151
// MI455X (gfx1250) — matrix operand order repaired and checked by reading, not compiled, not run on hardware
//
#include <hip/hip_runtime.h>
#include <stdint.h>

#define F 128
#define KSEL 1024
#define NEG_SLOPE 0.2f
#define EQCAP 2048

typedef __attribute__((ext_vector_type(16))) __bf16 v16bf;
typedef __attribute__((ext_vector_type(8)))  float v8f;

// 32-byte fragment; q[2] lets us fill it with two 16B LDS loads (ds_load_b128)
union Frag16 { v16bf v; unsigned short h[16]; uint4 q[2]; };

__device__ __forceinline__ unsigned short f2bf(float f) {
  unsigned u = __float_as_uint(f);
  u += 0x7FFFu + ((u >> 16) & 1u);   // round-to-nearest-even
  return (unsigned short)(u >> 16);
}

__device__ __forceinline__ void atomicMaxFloat(float* addr, float value) {
  int* ia = (int*)addr;
  int old = *ia;
  while (__int_as_float(old) < value) {
    int assumed = old;
    old = atomicCAS(ia, assumed, __float_as_int(value));
    if (old == assumed) break;
  }
}

__device__ __forceinline__ void edgeSD(const int* src, const int* dst, int E,
                                       int e, int& s, int& d) {
  if (e < E) { s = src[e]; d = dst[e]; }
  else       { s = e - E; d = s; }          // appended self loops
}

// ---------------------------------------------------------------------------
// WMMA GEMM:  C[nrows x 128] = A[nrows x 128] @ W[128 x 128]   (f32 in/out,
// bf16 WMMA, f32 accumulate). 256 thr = 8 waves; wave w owns a 16x128 strip.
//
// A tile: row-major bf16 in LDS. Per the ISA 16-bit A layout, lane l's 16
// halves are two contiguous 8-half runs (khi+[0,8) and khi+[16,24)) of its
// row -> two ds_load_b128.
// W tile: staged PRE-SWIZZLED so that lane l of column-tile c reads its 16
// halves from 32 contiguous bytes: sW[((ks*8+c)*32 + lane)*16 + i], where
// element W[k][n] maps to ks=k>>5, half=(k>>3)&1, i=(k&7)|(((k>>4)&1)<<3), c=n>>4,
// lane=half*16+(n&15)  -> two ds_load_b128, no repacking.
// ---------------------------------------------------------------------------
__global__ __launch_bounds__(256) void k_gemm128(const float* __restrict__ A,
                                                 const float* __restrict__ W,
                                                 float* __restrict__ C, int nrows) {
  __shared__ unsigned short sA[128 * 128];   // row-major bf16 bits
  __shared__ unsigned short sW[128 * 128];   // fragment-ready swizzle
  const int tid  = threadIdx.x;
  const int lane = tid & 31;
  const int wave = tid >> 5;
  const int rowBase = blockIdx.x * 128;

  for (int idx = tid; idx < 128 * 128; idx += 256) {
    int r = idx >> 7, c = idx & 127;
    int gr = rowBase + r;
    float av = (gr < nrows) ? A[(size_t)gr * F + c] : 0.0f;
    sA[idx] = f2bf(av);
    // swizzle W[k][n] -> fragment-ready position
    int k = idx >> 7, n = idx & 127;
    int ks = k >> 5, kw = k & 31;
    int half = (kw >> 3) & 1, i = (kw & 7) | ((kw >> 4) << 3);
    int ct = n >> 4, ll = n & 15;
    int pos = (((ks * 8 + ct) * 32 + half * 16 + ll) << 4) + i;
    sW[pos] = f2bf(W[idx]);
  }
  __syncthreads();

  v8f acc[8] = {};
  const int rl = wave * 16 + (lane & 15);
#pragma unroll
  for (int ks = 0; ks < 4; ++ks) {
    Frag16 a;
    const int khi = ks * 32 + ((lane >> 4) << 3);   // lanes 16-31 take K+8
    const unsigned short* ap = &sA[rl * 128 + khi];
    a.q[0] = *reinterpret_cast<const uint4*>(ap);        // halves khi..khi+7
    a.q[1] = *reinterpret_cast<const uint4*>(ap + 16);   // halves khi+16..+23
#pragma unroll
    for (int c = 0; c < 8; ++c) {
      Frag16 b;
      const unsigned short* bp = &sW[(((ks * 8 + c) * 32 + lane) << 4)];
      b.q[0] = *reinterpret_cast<const uint4*>(bp);
      b.q[1] = *reinterpret_cast<const uint4*>(bp + 8);
      acc[c] = __builtin_amdgcn_wmma_f32_16x16x32_bf16(
          false, a.v, false, b.v, (short)0, acc[c], false, false);
    }
  }

#pragma unroll
  for (int c = 0; c < 8; ++c) {
    const int col = c * 16 + (lane & 15);
#pragma unroll
    for (int d = 0; d < 8; ++d) {
      int row = rowBase + wave * 16 + ((lane >> 4) << 3) + d;
      if (row < nrows) C[(size_t)row * F + col] = acc[c][d];
    }
  }
}

// ---------------------------------------------------------------------------
// A2[1024x1024] = S^T (K x N) @ M (N x K).  Grid (8,8); block owns 128x128.
// K-dim = N (30000) streamed in 32-wide LDS-staged chunks (one WMMA k-step).
// sS: row-major per output-row m (m x 32)  -> A frags via 2x ds_load_b128.
// sM: fragment-ready swizzle ((c*32+lane)*16+i) -> B frags via 2x ds_load_b128.
// ---------------------------------------------------------------------------
__global__ __launch_bounds__(256) void k_gemm_atb(const float* __restrict__ S,
                                                  const float* __restrict__ M,
                                                  float* __restrict__ A2, int Kdim) {
  __shared__ unsigned short sS[128 * 32];    // sS[m*32+k] = S[k][mBase+m]
  __shared__ unsigned short sM[32 * 128];    // swizzled
  const int tid = threadIdx.x, lane = tid & 31, wave = tid >> 5;
  const int mBase = blockIdx.y * 128, nBase = blockIdx.x * 128;
  v8f acc[8] = {};
  const int rl = wave * 16 + (lane & 15);

  for (int kb = 0; kb < Kdim; kb += 32) {
    for (int idx = tid; idx < 128 * 32; idx += 256) {
      int m = idx >> 5, kk = idx & 31;
      int kg = kb + kk;
      sS[idx] = f2bf((kg < Kdim) ? S[(size_t)kg * KSEL + mBase + m] : 0.0f);
      int kk2 = idx >> 7, n = idx & 127;
      int kg2 = kb + kk2;
      float mv = (kg2 < Kdim) ? M[(size_t)kg2 * KSEL + nBase + n] : 0.0f;
      int half = (kk2 >> 3) & 1, i = (kk2 & 7) | ((kk2 >> 4) << 3);
      int ct = n >> 4, ll = n & 15;
      sM[((ct * 32 + half * 16 + ll) << 4) + i] = f2bf(mv);
    }
    __syncthreads();

    Frag16 a;
    const int khi = (lane >> 4) << 3;
    const unsigned short* ap = &sS[rl * 32 + khi];
    a.q[0] = *reinterpret_cast<const uint4*>(ap);
    a.q[1] = *reinterpret_cast<const uint4*>(ap + 16);
#pragma unroll
    for (int c = 0; c < 8; ++c) {
      Frag16 b;
      const unsigned short* bp = &sM[((c * 32 + lane) << 4)];
      b.q[0] = *reinterpret_cast<const uint4*>(bp);
      b.q[1] = *reinterpret_cast<const uint4*>(bp + 8);
      acc[c] = __builtin_amdgcn_wmma_f32_16x16x32_bf16(
          false, a.v, false, b.v, (short)0, acc[c], false, false);
    }
    __syncthreads();
  }

#pragma unroll
  for (int c = 0; c < 8; ++c) {
    const int col = nBase + c * 16 + (lane & 15);
#pragma unroll
    for (int d = 0; d < 8; ++d) {
      int row = mBase + wave * 16 + ((lane >> 4) << 3) + d;
      A2[(size_t)row * KSEL + col] = acc[c][d];
    }
  }
}

// ------------------------- element / edge kernels --------------------------
__global__ void k_fillf(float* p, float v, size_t n) {
  size_t i = (size_t)blockIdx.x * blockDim.x + threadIdx.x;
  if (i < n) p[i] = v;
}
__global__ void k_filli(int* p, int v, size_t n) {
  size_t i = (size_t)blockIdx.x * blockDim.x + threadIdx.x;
  if (i < n) p[i] = v;
}
__global__ void k_deg(const int* __restrict__ dst, float* deg, int E) {
  int e = blockIdx.x * blockDim.x + threadIdx.x;
  if (e < E) atomicAdd(&deg[dst[e]], 1.0f);
}
__global__ void k_findeg(float* deg, float* dinv, int n) {
  int i = blockIdx.x * blockDim.x + threadIdx.x;
  if (i < n) { float d = deg[i] + 1.0f; deg[i] = d; dinv[i] = rsqrtf(d); }
}
// out[i] = T[i]*dinv[i]^2 + bias  (GCN self-loop term)
__global__ void k_agg_init(const float* __restrict__ T, const float* __restrict__ dinv,
                           const float* __restrict__ bias, float* __restrict__ out, int n) {
  long long t = (long long)blockIdx.x * blockDim.x + threadIdx.x;
  if (t >= (long long)n * 32) return;
  int i = (int)(t >> 5), q = ((int)t & 31) * 4;
  float s = dinv[i] * dinv[i];
  const float* tv = T + (size_t)i * F + q;
  float* ov = out + (size_t)i * F + q;
  for (int j = 0; j < 4; ++j) ov[j] = tv[j] * s + bias[q + j];
}
__global__ void k_agg_edge(const float* __restrict__ T, const float* __restrict__ dinv,
                           const int* __restrict__ src, const int* __restrict__ dst,
                           float* __restrict__ out, int E) {
  long long t = (long long)blockIdx.x * blockDim.x + threadIdx.x;
  if (t >= (long long)E * 32) return;
  int e = (int)(t >> 5), q = ((int)t & 31) * 4;
  int s = src[e], d = dst[e];
  float nrm = dinv[s] * dinv[d];
  const float* tv = T + (size_t)s * F + q;
  float* ov = out + (size_t)d * F + q;
  for (int j = 0; j < 4; ++j) atomicAdd(&ov[j], tv[j] * nrm);
}
__global__ void k_relu(float* p, size_t n) {
  size_t i = (size_t)blockIdx.x * blockDim.x + threadIdx.x;
  if (i < n) p[i] = fmaxf(p[i], 0.0f);
}
// segment max of h[src] at dst (h >= 0 after relu, init 0, int-ordered atomicMax)
__global__ void k_segmax(const float* __restrict__ H, const int* __restrict__ src,
                         const int* __restrict__ dst, float* __restrict__ xq,
                         int E, int Et) {
  long long t = (long long)blockIdx.x * blockDim.x + threadIdx.x;
  if (t >= (long long)Et * 32) return;
  int e = (int)(t >> 5), q = ((int)t & 31) * 4;
  int s, d; edgeSD(src, dst, E, e, s, d);
  const float* hv = H + (size_t)s * F + q;
  int* ov = (int*)(xq + (size_t)d * F + q);
  for (int j = 0; j < 4; ++j) atomicMax(&ov[j], __float_as_int(hv[j]));
}
__global__ void k_dot128(const float* __restrict__ X, const float* __restrict__ bias,
                         const float* __restrict__ w, float* __restrict__ out, int n) {
  int i = blockIdx.x * blockDim.x + threadIdx.x;
  if (i >= n) return;
  const float* xv = X + (size_t)i * F;
  float acc = 0.0f;
  for (int f = 0; f < F; ++f) {
    float v = xv[f];
    if (bias) v += bias[f];
    acc += v * w[f];
  }
  out[i] = acc;
}
__global__ void k_score1(const float* __restrict__ qdot, const float* __restrict__ hdot,
                         const float* __restrict__ b_att, const int* __restrict__ src,
                         const int* __restrict__ dst, float* __restrict__ score,
                         float* __restrict__ smax, int E, int Et) {
  int e = blockIdx.x * blockDim.x + threadIdx.x;
  if (e >= Et) return;
  int s, d; edgeSD(src, dst, E, e, s, d);
  float v = qdot[d] + hdot[s] + b_att[0];
  v = (v > 0.0f) ? v : NEG_SLOPE * v;     // leaky relu
  score[e] = v;
  atomicMaxFloat(&smax[d], v);
}
__global__ void k_score2(float* __restrict__ score, const float* __restrict__ smax,
                         float* __restrict__ ssum, const int* __restrict__ src,
                         const int* __restrict__ dst, int E, int Et) {
  int e = blockIdx.x * blockDim.x + threadIdx.x;
  if (e >= Et) return;
  int s, d; edgeSD(src, dst, E, e, s, d);
  float ex = expf(score[e] - smax[d]);
  score[e] = ex;
  atomicAdd(&ssum[d], ex);
}
__global__ void k_score3(float* __restrict__ score, const float* __restrict__ ssum,
                         const int* __restrict__ src, const int* __restrict__ dst,
                         int E, int Et) {
  int e = blockIdx.x * blockDim.x + threadIdx.x;
  if (e >= Et) return;
  int s, d; edgeSD(src, dst, E, e, s, d);
  score[e] = score[e] / ssum[d];
}
__global__ void k_xc(const float* __restrict__ H, const float* __restrict__ score,
                     const int* __restrict__ src, const int* __restrict__ dst,
                     float* __restrict__ xc, int E, int Et) {
  long long t = (long long)blockIdx.x * blockDim.x + threadIdx.x;
  if (t >= (long long)Et * 32) return;
  int e = (int)(t >> 5), q = ((int)t & 31) * 4;
  int s, d; edgeSD(src, dst, E, e, s, d);
  float sc = score[e];
  const float* hv = H + (size_t)s * F + q;
  float* ov = xc + (size_t)d * F + q;
  for (int j = 0; j < 4; ++j) atomicAdd(&ov[j], hv[j] * sc);
}
__global__ void k_le(const float* __restrict__ xc, const float* __restrict__ w1,
                     const float* __restrict__ b1p, const float* __restrict__ w2,
                     const float* __restrict__ w3, const float* __restrict__ b3p,
                     float* aL, float* bL, float* cL, int n) {
  int i = blockIdx.x * blockDim.x + threadIdx.x;
  if (i >= n) return;
  const float* xv = xc + (size_t)i * F;
  float a = 0.0f, b = 0.0f, c = 0.0f;
  for (int f = 0; f < F; ++f) {
    float v = xv[f];
    a += v * w1[f]; b += v * w2[f]; c += v * w3[f];
  }
  aL[i] = a + b1p[0]; bL[i] = b; cL[i] = c + b3p[0];
}
__global__ void k_bsum(const float* __restrict__ bL, const int* __restrict__ src,
                       const int* __restrict__ dst, float* bsum, int E, int Et) {
  int e = blockIdx.x * blockDim.x + threadIdx.x;
  if (e >= Et) return;
  int s, d; edgeSD(src, dst, E, e, s, d);
  atomicAdd(&bsum[d], bL[s]);
}
__global__ void k_fit(const float* __restrict__ deg, const float* __restrict__ aL,
                      const float* __restrict__ bsum, const float* __restrict__ cL,
                      float* fit, unsigned* keys, int n) {
  int i = blockIdx.x * blockDim.x + threadIdx.x;
  if (i >= n) return;
  float agg = deg[i] * aL[i] - bsum[i] + cL[i];
  float f = 1.0f / (1.0f + expf(-agg));
  fit[i] = f;
  keys[i] = __float_as_uint(f);   // sigmoid in (0,1): bits are order-preserving
}

// ------------------------------ top-K (1 block) ----------------------------
__global__ __launch_bounds__(1024) void k_topk(const unsigned* __restrict__ keys, int n,
                                               int* __restrict__ perm_i,
                                               float* __restrict__ fit_top,
                                               float* __restrict__ permf) {
  __shared__ int sred[1024];
  __shared__ unsigned skey[1024];
  __shared__ int sidx[1024];
  __shared__ int eqIdx[EQCAP];
  __shared__ int s_cnt, s_eqn;
  const int tid = threadIdx.x;

  auto blockCount = [&](unsigned thrv) -> int {
    int c = 0;
    for (int i = tid; i < n; i += 1024) c += (keys[i] > thrv) ? 1 : 0;
    sred[tid] = c; __syncthreads();
    for (int s = 512; s > 0; s >>= 1) { if (tid < s) sred[tid] += sred[tid + s]; __syncthreads(); }
    int tot = sred[0]; __syncthreads();
    return tot;
  };

  unsigned lo = 0u, hi = 0xFFFFFFFFu;
  for (int it = 0; it < 33 && lo < hi; ++it) {
    unsigned mid = lo + ((hi - lo) >> 1);
    int c = blockCount(mid);
    if (c >= KSEL) lo = mid + 1u; else hi = mid;
  }
  const unsigned thr = lo;             // smallest T with count(>T) < K
  const int cg = blockCount(thr);      // strictly-greater count (< K)

  skey[tid] = 0u; sidx[tid] = 0x3FFFFFFF;
  if (tid == 0) { s_cnt = 0; s_eqn = 0; }
  __syncthreads();
  for (int i = tid; i < n; i += 1024) {
    unsigned u = keys[i];
    if (u > thr) {
      int p = atomicAdd(&s_cnt, 1);
      if (p < KSEL) { skey[p] = u; sidx[p] = i; }
    } else if (u == thr) {
      int p = atomicAdd(&s_eqn, 1);
      if (p < EQCAP) eqIdx[p] = i;
    }
  }
  __syncthreads();
  const int eqn = (s_eqn < EQCAP) ? s_eqn : EQCAP;
  const int need = KSEL - cg;          // ties taken in ascending index order
  for (int j = tid; j < eqn; j += 1024) {
    int v = eqIdx[j], r = 0;
    for (int l = 0; l < eqn; ++l) r += (eqIdx[l] < v) ? 1 : 0;
    if (r < need && (cg + r) < KSEL) { skey[cg + r] = thr; sidx[cg + r] = v; }
  }
  __syncthreads();

  // bitonic sort ascending on ((~key)<<32 | idx) -> key desc, idx asc (top_k order)
  for (unsigned kk = 2; kk <= 1024u; kk <<= 1) {
    for (unsigned j = kk >> 1; j > 0; j >>= 1) {
      unsigned i = (unsigned)tid, ixj = i ^ j;
      if (ixj > i) {
        unsigned long long a = (((unsigned long long)(~skey[i])) << 32) | (unsigned)sidx[i];
        unsigned long long b = (((unsigned long long)(~skey[ixj])) << 32) | (unsigned)sidx[ixj];
        bool asc = ((i & kk) == 0u);
        if ((a > b) == asc) {
          unsigned tk = skey[i]; skey[i] = skey[ixj]; skey[ixj] = tk;
          int ti = sidx[i]; sidx[i] = sidx[ixj]; sidx[ixj] = ti;
        }
      }
      __syncthreads();
    }
  }
  perm_i[tid]  = sidx[tid];
  fit_top[tid] = __uint_as_float(skey[tid]);
  permf[tid]   = (float)sidx[tid];
}

__global__ void k_mark(const int* __restrict__ perm_i, int* keep, int* colmap) {
  int t = blockIdx.x * blockDim.x + threadIdx.x;
  if (t < KSEL) { int p = perm_i[t]; keep[p] = 1; colmap[p] = t; }
}
__global__ void k_newx(const float* __restrict__ xc, const int* __restrict__ perm_i,
                       const float* __restrict__ fit_top, float* __restrict__ out) {
  long long t = (long long)blockIdx.x * blockDim.x + threadIdx.x;
  if (t >= (long long)KSEL * 32) return;
  int k = (int)(t >> 5), q = ((int)t & 31) * 4;
  int p = perm_i[k]; float f = fit_top[k];
  const float* xv = xc + (size_t)p * F + q;
  float* ov = out + (size_t)k * F + q;
  for (int j = 0; j < 4; ++j) ov[j] = xv[j] * f;
}
__global__ void k_sbuild(const float* __restrict__ score, const int* __restrict__ src,
                         const int* __restrict__ dst, const int* __restrict__ keep,
                         const int* __restrict__ colmap, float* __restrict__ S,
                         int E, int Et) {
  int e = blockIdx.x * blockDim.x + threadIdx.x;
  if (e >= Et) return;
  int s, d; edgeSD(src, dst, E, e, s, d);
  if (keep[d]) atomicAdd(&S[(size_t)s * KSEL + colmap[d]], score[e]);
}
__global__ void k_mbuild(const float* __restrict__ S, const int* __restrict__ src,
                         const int* __restrict__ dst, float* __restrict__ M,
                         int E, int Et) {
  long long t = (long long)blockIdx.x * blockDim.x + threadIdx.x;
  if (t >= (long long)Et * 256) return;
  int e = (int)(t >> 8), q = ((int)t & 255) * 4;
  int s, d; edgeSD(src, dst, E, e, s, d);
  const float* sv = S + (size_t)d * KSEL + q;
  float* mv = M + (size_t)s * KSEL + q;
  for (int j = 0; j < 4; ++j) atomicAdd(&mv[j], sv[j]);
}
__global__ void k_diag(float* A2) {
  int t = blockIdx.x * blockDim.x + threadIdx.x;
  if (t < KSEL) A2[(size_t)t * KSEL + t] = 1.0f;
}

// ---------------------------------------------------------------------------
extern "C" void kernel_launch(void* const* d_in, const int* in_sizes, int n_in,
                              void* d_out, int out_size, void* d_ws, size_t ws_size,
                              hipStream_t stream) {
  if (n_in < 15) return;
  const float* x     = (const float*)d_in[0];
  const int*   ei    = (const int*)d_in[1];
  const float* W1    = (const float*)d_in[2];
  const float* b1    = (const float*)d_in[3];
  const float* W2    = (const float*)d_in[4];
  const float* b2    = (const float*)d_in[5];
  const float* W_lin = (const float*)d_in[6];
  const float* b_lin = (const float*)d_in[7];
  const float* W_att = (const float*)d_in[8];
  const float* b_att = (const float*)d_in[9];
  const float* leW1  = (const float*)d_in[10];
  const float* leB1  = (const float*)d_in[11];
  const float* leW2  = (const float*)d_in[12];
  const float* leW3  = (const float*)d_in[13];
  const float* leB3  = (const float*)d_in[14];

  const int N  = in_sizes[0] / F;
  const int E  = in_sizes[1] / 2;
  const int Et = E + N;
  const int* src = ei;
  const int* dst = ei + E;

  // d_out layout: [new_x K*F][A2 K*K][fitness N][score Et][h N*F][perm K]
  float* out     = (float*)d_out;
  float* o_newx  = out;
  float* o_A2    = o_newx + (size_t)KSEL * F;
  float* o_fit   = o_A2 + (size_t)KSEL * KSEL;
  float* o_score = o_fit + N;
  float* o_h     = o_score + Et;
  float* o_perm  = o_h + (size_t)N * F;

  // workspace layout
  float* w = (float*)d_ws;
  float* bufT = w;  w += (size_t)N * F;         // GEMM out scratch
  float* bufH = w;  w += (size_t)N * F;         // h1 / xq_raw / xc
  float* deg  = w;  w += N;
  float* dinv = w;  w += N;
  float* qdot = w;  w += N;
  float* hdot = w;  w += N;
  float* smax = w;  w += N;
  float* ssum = w;  w += N;
  float* bsum = w;  w += N;
  float* aL   = w;  w += N;
  float* bL   = w;  w += N;
  float* cL   = w;  w += N;
  unsigned* keys = (unsigned*)w; w += N;
  float* fit_top = w; w += KSEL;
  int* perm_i = (int*)w; w += KSEL;
  int* keep   = (int*)w; w += N;
  int* colmap = (int*)w; w += N;
  float* Smat = w; w += (size_t)N * KSEL;
  float* Mmat = w; w += (size_t)N * KSEL;
  if ((size_t)((char*)w - (char*)d_ws) > ws_size) return;  // insufficient scratch

  auto nb = [](long long n) { return (unsigned)((n + 255) / 256); };

  // degrees (deg0+1 serves both gcn_norm and LEConv's self-loop degree)
  k_fillf<<<nb(N), 256, 0, stream>>>(deg, 0.f, (size_t)N);
  k_deg<<<nb(E), 256, 0, stream>>>(dst, deg, E);
  k_findeg<<<nb(N), 256, 0, stream>>>(deg, dinv, N);

  // GCN layer 1: h1 = relu(norm-agg(x@W1) + b1)
  k_gemm128<<<(N + 127) / 128, 256, 0, stream>>>(x, W1, bufT, N);
  k_agg_init<<<nb((long long)N * 32), 256, 0, stream>>>(bufT, dinv, b1, bufH, N);
  k_agg_edge<<<nb((long long)E * 32), 256, 0, stream>>>(bufT, dinv, src, dst, bufH, E);
  k_relu<<<nb((long long)N * F), 256, 0, stream>>>(bufH, (size_t)N * F);

  // GCN layer 2 -> h in d_out
  k_gemm128<<<(N + 127) / 128, 256, 0, stream>>>(bufH, W2, bufT, N);
  k_agg_init<<<nb((long long)N * 32), 256, 0, stream>>>(bufT, dinv, b2, o_h, N);
  k_agg_edge<<<nb((long long)E * 32), 256, 0, stream>>>(bufT, dinv, src, dst, o_h, E);
  k_relu<<<nb((long long)N * F), 256, 0, stream>>>(o_h, (size_t)N * F);

  // x_q = segment_max(h[src] @ dst) @ W_lin   (bufH reused for xq_raw)
  k_fillf<<<nb((long long)N * F), 256, 0, stream>>>(bufH, 0.f, (size_t)N * F);
  k_segmax<<<nb((long long)Et * 32), 256, 0, stream>>>(o_h, src, dst, bufH, E, Et);
  k_gemm128<<<(N + 127) / 128, 256, 0, stream>>>(bufH, W_lin, bufT, N);
  k_dot128<<<nb(N), 256, 0, stream>>>(bufT, b_lin, W_att, qdot, N);
  k_dot128<<<nb(N), 256, 0, stream>>>(o_h, (const float*)nullptr, W_att + F, hdot, N);

  // leaky-relu attention + segment softmax over dst  (score lives in d_out)
  k_fillf<<<nb(N), 256, 0, stream>>>(smax, -1e30f, (size_t)N);
  k_fillf<<<nb(N), 256, 0, stream>>>(ssum, 0.f, (size_t)N);
  k_score1<<<nb(Et), 256, 0, stream>>>(qdot, hdot, b_att, src, dst, o_score, smax, E, Et);
  k_score2<<<nb(Et), 256, 0, stream>>>(o_score, smax, ssum, src, dst, E, Et);
  k_score3<<<nb(Et), 256, 0, stream>>>(o_score, ssum, src, dst, E, Et);

  // xc (bufH reused)
  k_fillf<<<nb((long long)N * F), 256, 0, stream>>>(bufH, 0.f, (size_t)N * F);
  k_xc<<<nb((long long)Et * 32), 256, 0, stream>>>(o_h, o_score, src, dst, bufH, E, Et);

  // LEConv fitness -> sigmoid -> keys
  k_le<<<nb(N), 256, 0, stream>>>(bufH, leW1, leB1, leW2, leW3, leB3, aL, bL, cL, N);
  k_fillf<<<nb(N), 256, 0, stream>>>(bsum, 0.f, (size_t)N);
  k_bsum<<<nb(Et), 256, 0, stream>>>(bL, src, dst, bsum, E, Et);
  k_fit<<<nb(N), 256, 0, stream>>>(deg, aL, bsum, cL, o_fit, keys, N);

  // top-K select + order, mark kept clusters, new_x
  k_topk<<<1, 1024, 0, stream>>>(keys, N, perm_i, fit_top, o_perm);
  k_filli<<<nb(N), 256, 0, stream>>>(keep, 0, (size_t)N);
  k_filli<<<nb(N), 256, 0, stream>>>(colmap, 0, (size_t)N);
  k_mark<<<(KSEL + 255) / 256, 256, 0, stream>>>(perm_i, keep, colmap);
  k_newx<<<nb((long long)KSEL * 32), 256, 0, stream>>>(bufH, perm_i, fit_top, o_newx);

  // coarsening: S (scatter scores), M = A@S (scatter rows), A2 = S^T@M (WMMA)
  k_fillf<<<nb((long long)N * KSEL), 256, 0, stream>>>(Smat, 0.f, (size_t)N * KSEL);
  k_fillf<<<nb((long long)N * KSEL), 256, 0, stream>>>(Mmat, 0.f, (size_t)N * KSEL);
  k_sbuild<<<nb(Et), 256, 0, stream>>>(o_score, src, dst, keep, colmap, Smat, E, Et);
  k_mbuild<<<nb((long long)Et * 256), 256, 0, stream>>>(Smat, src, dst, Mmat, E, Et);
  dim3 g2(KSEL / 128, KSEL / 128);
  k_gemm_atb<<<g2, 256, 0, stream>>>(Smat, Mmat, o_A2, N);
  k_diag<<<(KSEL + 255) / 256, 256, 0, stream>>>(o_A2);
}
